// SelfAttentionHybrid_65481071407564
// MI455X (gfx1250) — matrix operand order repaired and checked by reading, not compiled, not run on hardware
//
#include <hip/hip_runtime.h>
#include <hip/hip_bf16.h>

typedef __attribute__((ext_vector_type(16))) _Float16 v16h;
typedef __attribute__((ext_vector_type(8)))  _Float16 v8h;
typedef __attribute__((ext_vector_type(8)))  float    v8f;
typedef __attribute__((ext_vector_type(4)))  float    v4f;

#define S_LEN 8192
#define E_DIM 64

__device__ __forceinline__ v16h cat16(v8h lo, v8h hi) {
  return __builtin_shufflevector(lo, hi, 0,1,2,3,4,5,6,7,8,9,10,11,12,13,14,15);
}
__device__ __forceinline__ v8f wmma_f16(v16h a, v16h b, v8f c) {
  // D = A(16x32,f16) * B(32x16,f16) + C(16x16,f32)
  return __builtin_amdgcn_wmma_f32_16x16x32_f16(false, a, false, b, (short)0, c, false, false);
}
__device__ __forceinline__ v8f v8f_splat(float v) {
  v8f r;
#pragma unroll
  for (int i = 0; i < 8; ++i) r[i] = v;
  return r;
}

// CDNA5 async copy global->LDS (ASYNCcnt-tracked, no VGPR round trip).
// addrspace(3) pointers are 32-bit LDS offsets; generic form is aperture|off,
// so truncating the generic pointer yields the LDS address operand.
__device__ __forceinline__ void async_b128(uint32_t lds_off, const _Float16* g) {
  asm volatile("global_load_async_to_lds_b128 %0, %1, off"
               :: "v"(lds_off), "v"(g)
               : "memory");
}
__device__ __forceinline__ void wait_async4() {
  asm volatile("s_wait_asynccnt 0x4" ::: "memory");
}
__device__ __forceinline__ void wait_async0() {
  asm volatile("s_wait_asynccnt 0x0" ::: "memory");
}

// ---------------------------------------------------------------------------
// k0: pack Wq/Wk/Wv (f32 64x64 row-major) into per-lane-contiguous f16
//     B-fragments (B layout: lane n=L%16, h=L/16, K = c*32 + 8h + p (p<8), c*32 + 16 + 8h + p-8 (p>=8)),
//     and zero the accumulators.
// ---------------------------------------------------------------------------
__global__ void k0_pack(const float* __restrict__ Wq, const float* __restrict__ Wk,
                        const float* __restrict__ Wv,
                        _Float16* __restrict__ Pq, _Float16* __restrict__ Pk,
                        _Float16* __restrict__ Pv,
                        float* __restrict__ acc, float* __restrict__ convsum) {
  int gid = blockIdx.x * blockDim.x + threadIdx.x;
  if (gid < 65) {
    if (gid < 64) acc[gid] = 0.f; else convsum[0] = 0.f;
  }
  if (gid >= 3 * 4096) return;
  int mat = gid / 4096, rem = gid % 4096;
  const float* W = (mat == 0) ? Wq : ((mat == 1) ? Wk : Wv);
  _Float16*    P = (mat == 0) ? Pq : ((mat == 1) ? Pk : Pv);
  int f = rem >> 9;          // frag index = t*2 + c  (0..7)
  int L = (rem >> 4) & 31;   // lane
  int p = rem & 15;          // position within v16h
  int c = f & 1, t = f >> 1;
  int n = L & 15, hh = L >> 4;
  int k = c * 32 + (p < 8 ? hh * 8 + p : 16 + hh * 8 + (p - 8));  // B layout: table order K per 32-chunk
  P[rem] = (_Float16)W[k * 64 + t * 16 + n];
}

// ---------------------------------------------------------------------------
// k1: QKV projection. One wave per 16 rows of x. 24 WMMAs per wave.
// ---------------------------------------------------------------------------
__global__ void __launch_bounds__(256) k1_qkv(
    const float* __restrict__ x,
    const _Float16* __restrict__ Pq, const _Float16* __restrict__ Pk,
    const _Float16* __restrict__ Pv,
    const float* __restrict__ bq, const float* __restrict__ bk,
    const float* __restrict__ bv,
    _Float16* __restrict__ Q, _Float16* __restrict__ K,
    _Float16* __restrict__ Vt) {
  const int lane = threadIdx.x & 31;
  const int wave = threadIdx.x >> 5;
  const int n = lane & 15, h = lane >> 4;
  const int row0 = (blockIdx.x * 8 + wave) * 16;

  // A-fragments of x (16x32 f16): p<8 -> K=c*32+8h+p ; p>=8 -> K=c*32+16+8h+(p-8)
  const float* xr = x + (row0 + n) * E_DIM;
  v16h a[2];
#pragma unroll
  for (int c = 0; c < 2; ++c) {
    v4f lo0 = *(const v4f*)(xr + c * 32 + 8 * h);
    v4f lo1 = *(const v4f*)(xr + c * 32 + 8 * h + 4);
    v4f hi0 = *(const v4f*)(xr + c * 32 + 16 + 8 * h);
    v4f hi1 = *(const v4f*)(xr + c * 32 + 16 + 8 * h + 4);
    v16h t;
#pragma unroll
    for (int i = 0; i < 4; ++i) {
      t[i]      = (_Float16)lo0[i];
      t[4 + i]  = (_Float16)lo1[i];
      t[8 + i]  = (_Float16)hi0[i];
      t[12 + i] = (_Float16)hi1[i];
    }
    a[c] = t;
  }

  const _Float16* Pm[3] = {Pq, Pk, Pv};
  const float*    bm[3] = {bq, bk, bv};
#pragma unroll
  for (int mat = 0; mat < 3; ++mat) {
#pragma unroll
    for (int t = 0; t < 4; ++t) {
      v8f cacc = v8f_splat(bm[mat][t * 16 + n]);  // bias splat (C col = n fixed)
#pragma unroll
      for (int c = 0; c < 2; ++c) {
        v16h bf = *(const v16h*)(Pm[mat] + ((t * 2 + c) * 32 + lane) * 16);
        cacc = wmma_f16(a[c], bf, cacc);
      }
      if (mat == 0) {        // Q, fold in softmax scale 1/sqrt(64)
#pragma unroll
        for (int i = 0; i < 8; ++i)
          Q[(row0 + i + 8 * h) * E_DIM + t * 16 + n] = (_Float16)(cacc[i] * 0.125f);
      } else if (mat == 1) { // K row-major
#pragma unroll
        for (int i = 0; i < 8; ++i)
          K[(row0 + i + 8 * h) * E_DIM + t * 16 + n] = (_Float16)cacc[i];
      } else {               // V transposed: Vt[col][token], contiguous 8-half store
        v8h pk;
#pragma unroll
        for (int i = 0; i < 8; ++i) pk[i] = (_Float16)cacc[i];
        *(v8h*)(Vt + (t * 16 + n) * S_LEN + row0 + 8 * h) = pk;
      }
    }
  }
}

// ---------------------------------------------------------------------------
// k2: conv prefilter: sum of sigmoid over 63 x 8191 VALID 2x2 conv outputs.
// ---------------------------------------------------------------------------
__global__ void k2_conv(const float* __restrict__ x, const float* __restrict__ cw,
                        const float* __restrict__ cb, float* __restrict__ convsum) {
  const int NS = 8191;
  const int total = 63 * 8191;
  const float w00 = cw[0], w01 = cw[1], w10 = cw[2], w11 = cw[3], b = cb[0];
  float part = 0.f;
  for (int idx = blockIdx.x * blockDim.x + threadIdx.x; idx < total;
       idx += gridDim.x * blockDim.x) {
    int e = idx / NS;        // 0..62
    int s = idx % NS;        // 0..8190
    const float* p0 = x + s * E_DIM + e;
    const float* p1 = x + (s + 1) * E_DIM + e;
    float logit = w00 * p0[0] + w10 * p0[1] + w01 * p1[0] + w11 * p1[1] + b;
    part += 1.f / (1.f + __expf(-logit));
  }
#pragma unroll
  for (int off = 16; off >= 1; off >>= 1) part += __shfl_xor(part, off, 32);
  if ((threadIdx.x & 31) == 0) atomicAdd(convsum, part);
}

// ---------------------------------------------------------------------------
// Stage one 64-key chunk of K (8KB, contiguous) and Vt (64 cols x 64 keys)
// into LDS with async b128 copies: 4 instructions per thread, ASYNCcnt += 4.
// ---------------------------------------------------------------------------
__device__ __forceinline__ void stage_chunk(const _Float16* __restrict__ K,
                                            const _Float16* __restrict__ Vt,
                                            _Float16* KsBuf, _Float16* VsBuf,
                                            int kb, int tid) {
#pragma unroll
  for (int s = 0; s < 2; ++s) {                 // K chunk: 512 x 16B segments
    int seg = tid + s * 256;
    async_b128((uint32_t)(uintptr_t)(KsBuf + seg * 8), K + kb * E_DIM + seg * 8);
  }
#pragma unroll
  for (int s = 0; s < 2; ++s) {                 // Vt chunk: 64 rows x 128B
    int sv = tid + s * 256;
    int col = sv >> 3, part = sv & 7;
    async_b128((uint32_t)(uintptr_t)(VsBuf + col * 64 + part * 8),
               Vt + col * S_LEN + kb + part * 8);
  }
}

// ---------------------------------------------------------------------------
// k3: flash attention, 64-key chunks double-buffered in LDS via
//     global_load_async_to_lds_b128. One wave = 16 queries; all 8 waves in a
//     block share the staged K/V (8x less L2 traffic). S^T trick: score C
//     layout maps element-wise onto the P A-fragment layout (no shuffles).
// ---------------------------------------------------------------------------
__global__ void __launch_bounds__(256) k3_attn(
    const _Float16* __restrict__ Q, const _Float16* __restrict__ K,
    const _Float16* __restrict__ Vt, float* __restrict__ acc) {
  __shared__ __align__(128) _Float16 Ks[2][64 * 64];   // 2 x 8KB
  __shared__ __align__(128) _Float16 Vs[2][64 * 64];   // 2 x 8KB
  const int tid  = threadIdx.x;
  const int lane = tid & 31;
  const int wave = tid >> 5;
  const int n = lane & 15, h = lane >> 4;
  const int q0 = (blockIdx.x * 8 + wave) * 16;

  // Q as B-fragments (E x 16 queries): per-lane 2x16B loads, table k order
  const _Float16* qrow = Q + (q0 + n) * E_DIM;
  const v16h bq0 = cat16(*(const v8h*)(qrow + 8 * h), *(const v8h*)(qrow + 16 + 8 * h));
  const v16h bq1 = cat16(*(const v8h*)(qrow + 32 + 8 * h), *(const v8h*)(qrow + 48 + 8 * h));

  v8f o[4];
#pragma unroll
  for (int t = 0; t < 4; ++t) o[t] = v8f_splat(0.f);
  float mrun = -1e30f, lrun = 0.f;

  stage_chunk(K, Vt, Ks[0], Vs[0], 0, tid);   // prologue: chunk 0 in flight

  for (int kb = 0, it = 0; kb < S_LEN; kb += 64, ++it) {
    const int cur = it & 1;
    const bool hasnext = (kb + 64) < S_LEN;
    if (hasnext) {
      stage_chunk(K, Vt, Ks[cur ^ 1], Vs[cur ^ 1], kb + 64, tid);
      wait_async4();   // in-order completion: drains the 4 loads of chunk `it`
    } else {
      wait_async0();
    }
    __syncthreads();   // all waves' async writes to buf[cur] are visible

    const _Float16* Kb = Ks[cur];
    const _Float16* Vb = Vs[cur];
    const v8f zero = v8f_splat(0.f);

    // S^T tiles: st[tt][i] = S[query n][key kb + 16*tt + i + 8h]
    v8f st[4];
#pragma unroll
    for (int tt = 0; tt < 4; ++tt) {
      const _Float16* kr = Kb + (tt * 16 + n) * E_DIM;
      v16h a0 = cat16(*(const v8h*)(kr + 8 * h), *(const v8h*)(kr + 16 + 8 * h));
      v16h a1 = cat16(*(const v8h*)(kr + 32 + 8 * h), *(const v8h*)(kr + 48 + 8 * h));
      v8f s = wmma_f16(a0, bq0, zero);
      st[tt] = wmma_f16(a1, bq1, s);
    }

    // online softmax over 64 keys (per-query stats live in lanes q and q+16)
    float mx = st[0][0];
#pragma unroll
    for (int tt = 0; tt < 4; ++tt)
#pragma unroll
      for (int i = 0; i < 8; ++i) mx = fmaxf(mx, st[tt][i]);
    mx = fmaxf(mx, __shfl_xor(mx, 16, 32));
    const float mnew = fmaxf(mrun, mx);
    const float r = __expf(mrun - mnew);
    mrun = mnew;

    v16h pa0, pa1;     // P A-fragments for keys 0..31 and 32..63 of the chunk
    float psum = 0.f;
#pragma unroll
    for (int i = 0; i < 8; ++i) {
      float p0 = __expf(st[0][i] - mnew); psum += p0; pa0[i]     = (_Float16)p0;
      float p1 = __expf(st[1][i] - mnew); psum += p1; pa0[8 + i] = (_Float16)p1;
      float p2 = __expf(st[2][i] - mnew); psum += p2; pa1[i]     = (_Float16)p2;
      float p3 = __expf(st[3][i] - mnew); psum += p3; pa1[8 + i] = (_Float16)p3;
    }
    lrun = lrun * r + (psum + __shfl_xor(psum, 16, 32));

    // rescale O rows (row = query i+8h; its factor lives in lane i+8h)
    float f[8];
#pragma unroll
    for (int i = 0; i < 8; ++i) f[i] = __shfl(r, i + 8 * h, 32);
#pragma unroll
    for (int t = 0; t < 4; ++t) {
#pragma unroll
      for (int i = 0; i < 8; ++i) o[t][i] *= f[i];
    }

    // O += P(16x64) * V(64x16 per column tile), V B-frags contiguous in LDS
#pragma unroll
    for (int t = 0; t < 4; ++t) {
      const _Float16* vr = Vb + (t * 16 + n) * 64;
      v16h b0 = cat16(*(const v8h*)(vr + 8 * h), *(const v8h*)(vr + 16 + 8 * h));
      v16h b1 = cat16(*(const v8h*)(vr + 32 + 8 * h), *(const v8h*)(vr + 48 + 8 * h));
      o[t] = wmma_f16(pa0, b0, o[t]);
      o[t] = wmma_f16(pa1, b1, o[t]);
    }
    __syncthreads();   // done reading buf[cur]; next iter may overwrite it
  }

  // finalize: divide rows by l, reduce over the 16 query rows -> column sums
  const float linv = 1.f / lrun;
  float lf[8];
#pragma unroll
  for (int i = 0; i < 8; ++i) lf[i] = __shfl(linv, i + 8 * h, 32);
#pragma unroll
  for (int t = 0; t < 4; ++t) {
    float cs = 0.f;
#pragma unroll
    for (int i = 0; i < 8; ++i) cs += o[t][i] * lf[i];
    cs += __shfl_xor(cs, 16, 32);
    if (lane < 16) atomicAdd(acc + t * 16 + lane, cs);
  }
}

// ---------------------------------------------------------------------------
// k4: out[j] = conv_feat * (1/S) * sum_e acc[e]*Wp[e,j] + bp[j]
// ---------------------------------------------------------------------------
__global__ void k4_final(const float* __restrict__ acc, const float* __restrict__ convsum,
                         const float* __restrict__ Wp, const float* __restrict__ bp,
                         float* __restrict__ out) {
  int j = threadIdx.x;
  if (j < 4) {
    float cf = convsum[0] / (63.f * 8191.f);
    float s = 0.f;
    for (int e = 0; e < E_DIM; ++e) s += acc[e] * Wp[e * 4 + j];
    out[j] = cf * s * (1.f / (float)S_LEN) + bp[j];
  }
}

extern "C" void kernel_launch(void* const* d_in, const int* in_sizes, int n_in,
                              void* d_out, int out_size, void* d_ws, size_t ws_size,
                              hipStream_t stream) {
  const float* x      = (const float*)d_in[0];
  const float* conv_w = (const float*)d_in[1];
  const float* conv_b = (const float*)d_in[2];
  const float* Wq     = (const float*)d_in[3];
  const float* bq     = (const float*)d_in[4];
  const float* Wk     = (const float*)d_in[5];
  const float* bk     = (const float*)d_in[6];
  const float* Wv     = (const float*)d_in[7];
  const float* bv     = (const float*)d_in[8];
  const float* Wp     = (const float*)d_in[9];
  const float* bp     = (const float*)d_in[10];
  float* out = (float*)d_out;

  char* ws = (char*)d_ws;
  _Float16* Q   = (_Float16*)(ws);                       // 1 MB
  _Float16* K   = (_Float16*)(ws + 1048576);             // 1 MB
  _Float16* Vt  = (_Float16*)(ws + 2097152);             // 1 MB (transposed 64x8192)
  _Float16* Pq  = (_Float16*)(ws + 3145728);             // 8 KB packed Wq
  _Float16* Pk  = (_Float16*)(ws + 3145728 + 8192);      // 8 KB packed Wk
  _Float16* Pv  = (_Float16*)(ws + 3145728 + 16384);     // 8 KB packed Wv
  float* acc     = (float*)(ws + 3145728 + 24576);       // 64 f32 column sums
  float* convsum = (float*)(ws + 3145728 + 24576 + 256); // 1 f32

  k0_pack<<<48, 256, 0, stream>>>(Wq, Wk, Wv, Pq, Pk, Pv, acc, convsum);
  k1_qkv<<<64, 256, 0, stream>>>(x, Pq, Pk, Pv, bq, bk, bv, Q, K, Vt);
  k2_conv<<<256, 256, 0, stream>>>(x, conv_w, conv_b, convsum);
  k3_attn<<<64, 256, 0, stream>>>(Q, K, Vt, acc);
  k4_final<<<1, 32, 0, stream>>>(acc, convsum, Wp, bp, out);
}
